// HHGraphSum_5574867550777
// MI455X (gfx1250) — matrix operand order repaired and checked by reading, not compiled, not run on hardware
//
#include <hip/hip_runtime.h>
#include <math.h>

// ---------------- problem constants ----------------
static const int cNW = 100000, cNS = 3200;
static const int cEWS = 300000, cEWW = 300000, cESS = 100000;

// ---------------- WMMA plumbing ----------------
typedef __attribute__((ext_vector_type(16))) _Float16 v16h;
typedef __attribute__((ext_vector_type(8)))  _Float16 v8h;
typedef __attribute__((ext_vector_type(8)))  float    v8f;

#define DEVI static __device__ __forceinline__

DEVI float sigf(float x) { return 1.f / (1.f + __expf(-x)); }

// force a wave-uniform condition into a scalar branch (no EXEC churn)
DEVI bool uniform(bool c) { return __builtin_amdgcn_readfirstlane((int)c) != 0; }

DEVI v8f wmma_f16(v16h a, v16h b, v8f c) {
  return __builtin_amdgcn_wmma_f32_16x16x32_f16(false, a, false, b, (short)0, c, false, false);
}

// ---- A fragment (16x32) from row-major f32 [M x K] ----
// lane L: m = row0 + L%16 ; element e: K = k0 + 8*(L/16) + (e<8 ? e : e+8)
DEVI v16h afrag_f32_fast(const float* A, int lda, int row0, int k0, int lane) {
  const float* p = A + (size_t)(row0 + (lane & 15)) * lda + k0 + ((lane >> 4) << 3);
  v16h a;
#pragma unroll
  for (int j = 0; j < 4; ++j) {
    float2 t0 = *(const float2*)(p + 2 * j);        // run 0: 8 contiguous floats
    float2 t1 = *(const float2*)(p + 16 + 2 * j);   // run 1: 8 contiguous floats
    a[2 * j]     = (_Float16)t0.x; a[2 * j + 1]     = (_Float16)t0.y;
    a[8 + 2 * j] = (_Float16)t1.x; a[8 + 2 * j + 1] = (_Float16)t1.y;
  }
  return a;
}

DEVI v16h afrag_f32(const float* A, int lda, int row0, int k0, int M, int K, int lane) {
  if (uniform(row0 + 16 <= M && k0 + 32 <= K))      // scalar branch
    return afrag_f32_fast(A, lda, row0, k0, lane);
  int m  = row0 + (lane & 15);
  int kb = k0 + ((lane >> 4) << 3);
  int mc = m < M ? m : M - 1;
  v16h a;
#pragma unroll
  for (int e = 0; e < 16; ++e) {
    int kk  = kb + (e < 8 ? e : e + 8);
    int kkc = kk < K ? kk : K - 1;
    float v = A[(size_t)mc * lda + kkc];            // unconditional clamped load
    a[e] = (_Float16)((m < M && kk < K) ? v : 0.f); // masked via select
  }
  return a;
}

// ---- A fragment from f16 LDS (caller guarantees range + 16B alignment): two ds_load_b128 ----
DEVI v16h afrag_f16lds(const _Float16* A, int lda, int row0, int k0, int lane) {
  const _Float16* p = A + (row0 + (lane & 15)) * lda + k0 + ((lane >> 4) << 3);
  v8h lo = *(const v8h*)p;
  v8h hi = *(const v8h*)(p + 16);
  v16h a;
#pragma unroll
  for (int e = 0; e < 8; ++e) { a[e] = lo[e]; a[e + 8] = hi[e]; }
  return a;
}

// ---- B fragment (32x16) from row-major f32 [K x N] ----
// lane L: n = col0 + L%16 ; element e: K = k0 + 8*(L/16) + (e<8 ? e : e+8)
DEVI v16h bfrag_f32(const float* B, int ldb, int k0, int col0, int K, int N, int lane) {
  int n  = col0 + (lane & 15);
  int kb = k0 + ((lane >> 4) << 3);
  if (uniform(col0 + 16 <= N && k0 + 32 <= K)) {    // scalar branch
    v16h b;
#pragma unroll
    for (int e = 0; e < 16; ++e) b[e] = (_Float16)B[(size_t)(kb + (e < 8 ? e : e + 8)) * ldb + n];
    return b;
  }
  int nc = n < N ? n : N - 1;
  v16h b;
#pragma unroll
  for (int e = 0; e < 16; ++e) {
    int kk  = kb + (e < 8 ? e : e + 8);
    int kkc = kk < K ? kk : K - 1;
    float v = B[(size_t)kkc * ldb + nc];
    b[e] = (_Float16)((n < N && kk < K) ? v : 0.f);
  }
  return b;
}

// f16 B fragment, caller guarantees in-range (used for LSTM whh preload)
DEVI v16h bfrag_f16_nochk(const _Float16* B, int ldb, int k0, int col0, int lane) {
  int n  = col0 + (lane & 15);
  int kb = k0 + ((lane >> 4) << 3);
  v16h b;
#pragma unroll
  for (int e = 0; e < 16; ++e) b[e] = B[(size_t)(kb + (e < 8 ? e : e + 8)) * ldb + n];
  return b;
}

// B = Bt^T where Bt is row-major [N x K] (Q·K^T): two runs of 8 per lane
DEVI v16h bfragT_f32(const float* Bt, int ldbt, int k0, int col0, int K, int N, int lane) {
  int n  = col0 + (lane & 15);
  int kb = k0 + ((lane >> 4) << 3);
  if (uniform(col0 + 16 <= N && k0 + 32 <= K)) {
    const float* p = Bt + (size_t)n * ldbt + kb;    // runs at +0 and +16, 8 floats each
    v16h b;
#pragma unroll
    for (int j = 0; j < 8; ++j) {
      float2 t = *(const float2*)(p + (j < 4 ? 2 * j : 2 * j + 8));
      b[2 * j] = (_Float16)t.x; b[2 * j + 1] = (_Float16)t.y;
    }
    return b;
  }
  int nc = n < N ? n : N - 1;
  v16h b;
#pragma unroll
  for (int e = 0; e < 16; ++e) {
    int kk  = kb + (e < 8 ? e : e + 8);
    int kkc = kk < K ? kk : K - 1;
    float v = Bt[(size_t)nc * ldbt + kkc];
    b[e] = (_Float16)((n < N && kk < K) ? v : 0.f);
  }
  return b;
}

// ---------------- generic WMMA GEMM ----------------
// C[M,N] = act(A[M,K] @ B[K,N] + bias + res). Block=128: 4 waves, each a 32x32 tile.
#define ACT_NONE 0
#define ACT_RELU 1
#define ACT_SIG  2

template <int ACT>
DEVI void store_tile(v8f acc, const float* bias, const float* res, float* C, int ldc,
                     int row0, int col0, int M, int N, int lane) {
  int n = col0 + (lane & 15);
  if (n >= N) return;
  float bv = bias ? bias[n] : 0.f;
#pragma unroll
  for (int r = 0; r < 8; ++r) {
    int m = row0 + r + ((lane >> 4) << 3);
    if (m >= M) continue;
    float v = acc[r] + bv;
    if (res) v += res[(size_t)m * ldc + n];
    if (ACT == ACT_RELU) v = fmaxf(v, 0.f);
    if (ACT == ACT_SIG)  v = sigf(v);
    C[(size_t)m * ldc + n] = v;
  }
}

template <int ACT>
__global__ void gemm_k(const float* __restrict__ A, int lda,
                       const float* __restrict__ B, int ldb,
                       const float* __restrict__ bias,
                       const float* __restrict__ res,
                       float* __restrict__ C, int ldc,
                       int M, int N, int K) {
  int lane = threadIdx.x & 31;
  int wv   = threadIdx.x >> 5;
  int col0 = (blockIdx.x * 4 + wv) * 32;
  int row0 = blockIdx.y * 32;
  if (uniform(row0 >= M || col0 >= N)) return;   // wave-uniform exit
  v8f a00 = {}, a01 = {}, a10 = {}, a11 = {};
  for (int k0 = 0; k0 < K; k0 += 32) {
    __builtin_prefetch(&A[(size_t)(row0 + (lane & 15)) * lda + k0 + 32], 0, 3);
    v16h af0 = afrag_f32(A, lda, row0,      k0, M, K, lane);
    v16h af1 = afrag_f32(A, lda, row0 + 16, k0, M, K, lane);
    v16h bf0 = bfrag_f32(B, ldb, k0, col0,      K, N, lane);
    v16h bf1 = bfrag_f32(B, ldb, k0, col0 + 16, K, N, lane);
    a00 = wmma_f16(af0, bf0, a00);
    a01 = wmma_f16(af0, bf1, a01);
    a10 = wmma_f16(af1, bf0, a10);
    a11 = wmma_f16(af1, bf1, a11);
  }
  store_tile<ACT>(a00, bias, res, C, ldc, row0,      col0,      M, N, lane);
  store_tile<ACT>(a01, bias, res, C, ldc, row0,      col0 + 16, M, N, lane);
  store_tile<ACT>(a10, bias, res, C, ldc, row0 + 16, col0,      M, N, lane);
  store_tile<ACT>(a11, bias, res, C, ldc, row0 + 16, col0 + 16, M, N, lane);
}

static void gemm(hipStream_t st, const float* A, int lda, const float* B, int ldb,
                 const float* bias, const float* res, float* C, int ldc,
                 int M, int N, int K, int act) {
  dim3 g((N + 127) / 128, (M + 31) / 32), b(128);
  if (act == ACT_RELU)     gemm_k<ACT_RELU><<<g, b, 0, st>>>(A, lda, B, ldb, bias, res, C, ldc, M, N, K);
  else if (act == ACT_SIG) gemm_k<ACT_SIG ><<<g, b, 0, st>>>(A, lda, B, ldb, bias, res, C, ldc, M, N, K);
  else                     gemm_k<ACT_NONE><<<g, b, 0, st>>>(A, lda, B, ldb, bias, res, C, ldc, M, N, K);
}

// ---------------- misc elementwise kernels ----------------
__global__ void sinusoid_k(float* tab, int n, int d, int zero_first) {
  int i = blockIdx.x * blockDim.x + threadIdx.x;
  if (i >= n * d) return;
  int p = i / d, c = i % d;
  float ang = (float)p / __powf(10000.f, (float)(2 * (c / 2)) / (float)d);
  float v = (c & 1) ? __cosf(ang) : __sinf(ang);
  if (zero_first && p == 0) v = 0.f;
  tab[i] = v;
}

__global__ void gather_rows_k(const float* __restrict__ tab, const int* __restrict__ idx,
                              float* __restrict__ out, int n, int d) {
  int i = blockIdx.x * blockDim.x + threadIdx.x;
  if (i >= n * d) return;
  int r = i / d, c = i % d;
  out[i] = tab[(size_t)idx[r] * d + c];
}

__global__ void add_gather_pos_k(const float* x, const float* tab, const int* pos,
                                 float* y, int n, int d) {
  int i = blockIdx.x * blockDim.x + threadIdx.x;
  if (i >= n * d) return;
  int r = i / d, c = i % d;
  y[i] = x[i] + tab[pos[r] * d + c];
}

__global__ void add_pos_mod_k(float* x, const float* tab, int n, int d, int period) {
  int i = blockIdx.x * blockDim.x + threadIdx.x;
  if (i >= n * d) return;
  int r = i / d, c = i % d;
  x[i] += tab[(r % period) * d + c];
}

__global__ void f32_to_f16_k(const float* __restrict__ x, _Float16* __restrict__ y, int n) {
  int i = blockIdx.x * blockDim.x + threadIdx.x;
  if (i < n) y[i] = (_Float16)x[i];
}

__global__ void fill_u32_k(unsigned* p, unsigned v, int n) {
  int i = blockIdx.x * blockDim.x + threadIdx.x;
  if (i < n) p[i] = v;
}

__global__ void elu_k(float* x, int n) {
  int i = blockIdx.x * blockDim.x + threadIdx.x;
  if (i < n) { float v = x[i]; x[i] = v > 0.f ? v : (__expf(v) - 1.f); }
}

// row-wise LayerNorm (wave per row); y = LN(x (+res)) * g + b ; in-place safe
__global__ void ln_k(const float* __restrict__ x, const float* __restrict__ res,
                     const float* __restrict__ g, const float* __restrict__ b,
                     float* __restrict__ y, int n, int d) {
  int row = blockIdx.x * (blockDim.x >> 5) + (threadIdx.x >> 5);
  if (row >= n) return;
  int lane = threadIdx.x & 31;
  const float* xr = x + (size_t)row * d;
  const float* rr = res ? res + (size_t)row * d : nullptr;
  float s = 0.f;
  for (int c = lane; c < d; c += 32) s += xr[c] + (rr ? rr[c] : 0.f);
  for (int o = 16; o > 0; o >>= 1) s += __shfl_xor(s, o, 32);
  float mean = s / d;
  float vs = 0.f;
  for (int c = lane; c < d; c += 32) {
    float v = xr[c] + (rr ? rr[c] : 0.f) - mean; vs += v * v;
  }
  for (int o = 16; o > 0; o >>= 1) vs += __shfl_xor(vs, o, 32);
  float rstd = rsqrtf(vs / d + 1e-5f);
  float* yr = y + (size_t)row * d;
  for (int c = lane; c < d; c += 32)
    yr[c] = (xr[c] + (rr ? rr[c] : 0.f) - mean) * rstd * g[c] + b[c];
}

// ---------------- text-CNN: per-tap GEMM decomposition ----------------
// y[t,f] = relu( sum_dk sum_c we[t+dk-pad, c] * w[dk,c,f] + b[f] ) then max over t.
// Staged tile is zero-padded: 8 guard rows above, zero rows below, channels 300->320,
// so A-fragments are unconditional 16B-aligned LDS vector loads (row stride 640B).
struct ConvArgs { const float* w[6]; const float* b[6]; };

#define CROWS 80
#define CLDA  320   // padded channel stride (halves); 320*2 = 640 B, 16B-aligned

__global__ void conv_ngram_k(const int* __restrict__ sent_words,
                             const float* __restrict__ embed,
                             ConvArgs ca, float* __restrict__ ngram) {
  __shared__ _Float16 wef[CROWS * CLDA];   // 50 KB
  __shared__ int swords[64];
  int s = blockIdx.x, ki = blockIdx.y, k = ki + 2;
  int tid = threadIdx.x;

  // async copy of this sentence's word ids: global -> LDS (ASYNCcnt-tracked)
  if (tid < 50) {
    unsigned ldsoff = (unsigned)(size_t)&swords[tid];
    unsigned long long gaddr = (unsigned long long)(size_t)(sent_words + s * 50 + tid);
    asm volatile("global_load_async_to_lds_b32 %0, %1, off\n\t"
                 "s_wait_asynccnt 0x0"
                 :: "v"(ldsoff), "v"(gaddr) : "memory");
  }
  // zero the whole padded tile (as u32 pairs)
  {
    unsigned* wz = (unsigned*)wef;
    for (int i = tid; i < CROWS * CLDA / 2; i += 128) wz[i] = 0u;
  }
  __syncthreads();
  // fill valid rows 8..57 (t' = 0..49), channels 0..299
  for (int i = tid; i < 50 * 300; i += 128) {
    int t = i / 300, c = i % 300;
    wef[(t + 8) * CLDA + c] = (_Float16)embed[(size_t)swords[t] * 300 + c];
  }
  __syncthreads();

  int lane = tid & 31, wv = tid >> 5;
  int pad = (k - 1) / 2;
  const float* W  = ca.w[ki];
  const float* Bb = ca.b[ki];
  int n0 = wv * 16;                        // FEAT tile for this wave
  v8f acc[4] = {v8f{}, v8f{}, v8f{}, v8f{}};
  for (int dk = 0; dk < k; ++dk) {
    const float* Wd = W + (size_t)dk * 300 * 50;   // [300 x 50] slice for this tap
    int rbase = dk - pad + 8;                      // shifted, padded row origin
    for (int k0 = 0; k0 < CLDA; k0 += 32) {        // 10 chunks; cols 300..319 are zero in A
      v16h b = bfrag_f32(Wd, 50, k0, n0, 300, 50, lane);
      v16h a0 = afrag_f16lds(wef, CLDA, 0 * 16 + rbase, k0, lane);   // batch all LDS loads
      v16h a1 = afrag_f16lds(wef, CLDA, 1 * 16 + rbase, k0, lane);
      v16h a2 = afrag_f16lds(wef, CLDA, 2 * 16 + rbase, k0, lane);
      v16h a3 = afrag_f16lds(wef, CLDA, 3 * 16 + rbase, k0, lane);
      acc[0] = wmma_f16(a0, b, acc[0]);
      acc[1] = wmma_f16(a1, b, acc[1]);
      acc[2] = wmma_f16(a2, b, acc[2]);
      acc[3] = wmma_f16(a3, b, acc[3]);
    }
  }
  int f = n0 + (lane & 15);
  float bv = (f < 50) ? Bb[f] : 0.f;
  float colmax = 0.f;                      // relu >= 0, so 0 is the exact identity
#pragma unroll
  for (int mt = 0; mt < 4; ++mt) {
#pragma unroll
    for (int r = 0; r < 8; ++r) {
      int t = mt * 16 + r + ((lane >> 4) << 3);
      float v = fmaxf(acc[mt][r] + bv, 0.f);
      if (t >= 50) v = 0.f;
      colmax = fmaxf(colmax, v);
    }
  }
  colmax = fmaxf(colmax, __shfl_xor(colmax, 16, 32));
  if (lane < 16 && f < 50) ngram[(size_t)s * 300 + ki * 50 + f] = colmax;
}

// ---------------- fused LSTM sequence (one persistent block, WMMA gate GEMM) ----------------
// xW[B*T,512] holds x@wih + b.  whh_h: f16 [128,512] (loop-invariant; fragments preloaded).
__global__ void lstm_seq_k(const float* __restrict__ xW,
                           const _Float16* __restrict__ whh_h,
                           float* __restrict__ hsout, int ldh, int colOff, int reverse) {
  extern __shared__ char smem[];
  float*    g   = (float*)smem;                    // 64*512 f32
  float*    cst = g + 64 * 512;                    // 64*128 f32
  _Float16* hh  = (_Float16*)(cst + 64 * 128);     // 64*128 f16
  int tid = threadIdx.x, lane = tid & 31, wv = tid >> 5;  // 1024 thr = 32 waves
  int col0 = wv * 16;                              // wave owns one of 32 N-tiles
  v16h Bf[4];
#pragma unroll
  for (int kc = 0; kc < 4; ++kc) Bf[kc] = bfrag_f16_nochk(whh_h, 512, kc * 32, col0, lane);
  for (int i = tid; i < 64 * 128; i += 1024) { cst[i] = 0.f; hh[i] = (_Float16)0.f; }
  __syncthreads();
  for (int t = 0; t < 50; ++t) {
    int ts = reverse ? 49 - t : t;
#pragma unroll
    for (int mt = 0; mt < 4; ++mt) {
      int row0 = mt * 16;
      // batch the four LDS A-fragment loads, then issue the WMMA chain
      v16h a0 = afrag_f16lds(hh, 128, row0, 0 * 32, lane);
      v16h a1 = afrag_f16lds(hh, 128, row0, 1 * 32, lane);
      v16h a2 = afrag_f16lds(hh, 128, row0, 2 * 32, lane);
      v16h a3 = afrag_f16lds(hh, 128, row0, 3 * 32, lane);
      v8f acc = {};
      acc = wmma_f16(a0, Bf[0], acc);
      acc = wmma_f16(a1, Bf[1], acc);
      acc = wmma_f16(a2, Bf[2], acc);
      acc = wmma_f16(a3, Bf[3], acc);
      int n = col0 + (lane & 15);
#pragma unroll
      for (int r = 0; r < 8; ++r) {
        int m = row0 + r + ((lane >> 4) << 3);
        g[m * 512 + n] = acc[r] + xW[(size_t)(m * 50 + ts) * 512 + n];
      }
    }
    __syncthreads();
    for (int i = tid; i < 64 * 128; i += 1024) {
      int m = i >> 7, c = i & 127;
      float gi = g[m * 512 + c],       gf = g[m * 512 + 128 + c];
      float gg = g[m * 512 + 256 + c], go = g[m * 512 + 384 + c];
      float cc = sigf(gf) * cst[i] + sigf(gi) * tanhf(gg);
      float h  = sigf(go) * tanhf(cc);
      cst[i] = cc; hh[i] = (_Float16)h;
      hsout[(size_t)(m * 50 + ts) * ldh + colOff + c] = h;
    }
    __syncthreads();
  }
}

// ---------------- transformer attention (per batch*head block) ----------------
__global__ void attn_k(const float* __restrict__ q, const float* __restrict__ k,
                       const float* __restrict__ v, float* __restrict__ o) {
  __shared__ float S[64 * 64];
  int bh = blockIdx.x, b = bh >> 2, h = bh & 3;
  int tid = threadIdx.x, lane = tid & 31, wv = tid >> 5;
  const float* qb = q + (size_t)(b * 50) * 128 + h * 32;
  const float* kb = k + (size_t)(b * 50) * 128 + h * 32;
  const float* vb = v + (size_t)(b * 50) * 128 + h * 32;
  int row0 = wv * 16;
  v16h aq = afrag_f32(qb, 128, row0, 0, 50, 32, lane);   // Q fragment reused over 4 col tiles
  for (int nt = 0; nt < 4; ++nt) {
    int col0 = nt * 16;
    v16h bf = bfragT_f32(kb, 128, 0, col0, 32, 50, lane);
    v8f acc = {};
    acc = wmma_f16(aq, bf, acc);
    int n = col0 + (lane & 15);
    const float scale = 0.1767766952966369f;  // 1/sqrt(32)
#pragma unroll
    for (int r = 0; r < 8; ++r)
      S[(row0 + r + ((lane >> 4) << 3)) * 64 + n] = acc[r] * scale;
  }
  __syncthreads();
  for (int rr = tid; rr < 64; rr += blockDim.x) {
    if (rr < 50) {
      float mx = -1e30f;
      for (int c = 0; c < 50; ++c) mx = fmaxf(mx, S[rr * 64 + c]);
      float sm = 0.f;
      for (int c = 0; c < 50; ++c) { float e = __expf(S[rr * 64 + c] - mx); S[rr * 64 + c] = e; sm += e; }
      float inv = 1.f / sm;
      for (int c = 0; c < 50; ++c) S[rr * 64 + c] *= inv;
      for (int c = 50; c < 64; ++c) S[rr * 64 + c] = 0.f;
    } else {
      for (int c = 0; c < 64; ++c) S[rr * 64 + c] = 0.f;
    }
  }
  __syncthreads();
  for (int nt = 0; nt < 2; ++nt) {
    int col0 = nt * 16;
    v8f acc = {};
#pragma unroll
    for (int kc = 0; kc < 2; ++kc) {
      int m = row0 + (lane & 15);
      int kb2 = kc * 32 + ((lane >> 4) << 3);
      const float4* sp0 = (const float4*)&S[m * 64 + kb2];        // ds_load_b128 x2
      const float4* sp1 = (const float4*)&S[m * 64 + kb2 + 16];
      float4 s0 = sp0[0], s1 = sp0[1], s2 = sp1[0], s3 = sp1[1];
      v16h a;
      a[0]=(_Float16)s0.x; a[1]=(_Float16)s0.y; a[2]=(_Float16)s0.z; a[3]=(_Float16)s0.w;
      a[4]=(_Float16)s1.x; a[5]=(_Float16)s1.y; a[6]=(_Float16)s1.z; a[7]=(_Float16)s1.w;
      a[8]=(_Float16)s2.x; a[9]=(_Float16)s2.y; a[10]=(_Float16)s2.z; a[11]=(_Float16)s2.w;
      a[12]=(_Float16)s3.x; a[13]=(_Float16)s3.y; a[14]=(_Float16)s3.z; a[15]=(_Float16)s3.w;
      v16h bf2 = bfrag_f32(vb, 128, kc * 32, col0, 50, 32, lane);
      acc = wmma_f16(a, bf2, acc);
    }
    int n = col0 + (lane & 15);
#pragma unroll
    for (int r = 0; r < 8; ++r) {
      int m = row0 + r + ((lane >> 4) << 3);
      if (m < 50) o[(size_t)(b * 50 + m) * 128 + h * 32 + n] = acc[r];
    }
  }
}

// ---------------- GAT kernels ----------------
__global__ void fold_attw_k(const float* __restrict__ w, const float* __restrict__ av,
                            float* __restrict__ outp, int din, int out, int H) {
  int i = blockIdx.x * blockDim.x + threadIdx.x;
  if (i >= din * H) return;
  int c = i / H, h = i % H, dh = out / H;
  float s = 0.f;
  for (int d = 0; d < dh; ++d) s += w[(size_t)c * out + h * dh + d] * av[h * dh + d];
  outp[i] = s;
}

__global__ void zea_k(const float* tftab, const float* we, const float* ae,
                      float* zea, int out, int H) {
  int i = blockIdx.x * blockDim.x + threadIdx.x;
  if (i >= 10 * H) return;
  int r = i / H, h = i % H, dh = out / H;
  float s = 0.f;
  for (int d = 0; d < dh; ++d) {
    float z = 0.f;
    for (int c = 0; c < 50; ++c) z += tftab[r * 50 + c] * we[(size_t)c * out + h * dh + d];
    s += z * ae[h * dh + d];
  }
  zea[i] = s;
}

DEVI unsigned fkey(float f) { int i = __float_as_int(f); return (i >= 0) ? ((unsigned)i ^ 0x80000000u) : ~(unsigned)i; }
DEVI float funkey(unsigned u) { int i = (u & 0x80000000u) ? (int)(u ^ 0x80000000u) : (int)~u; return __int_as_float(i); }

__global__ void edge_score_k(const int* esrc, const int* edst, const int* etf,
                             const float* als, const float* ard, const float* zea,
                             float* sbuf, unsigned* mbuf, int E, int H) {
  int i = blockIdx.x * blockDim.x + threadIdx.x;
  if (i >= E * H) return;
  int e = i / H, h = i % H;
  float s = als[(size_t)esrc[e] * H + h] + ard[(size_t)edst[e] * H + h] + zea[etf[e] * H + h];
  s = (s > 0.f) ? s : 0.2f * s;   // leaky_relu(0.2)
  sbuf[i] = s;
  atomicMax(&mbuf[(size_t)edst[e] * H + h], fkey(s));
}

__global__ void edge_exp_k(const int* edst, const unsigned* mbuf, float* sbuf,
                           float* den, int E, int H) {
  int i = blockIdx.x * blockDim.x + threadIdx.x;
  if (i >= E * H) return;
  int e = i / H, h = i % H;
  float m = funkey(mbuf[(size_t)edst[e] * H + h]);
  float v = __expf(sbuf[i] - m);
  sbuf[i] = v;
  atomicAdd(&den[(size_t)edst[e] * H + h], v);
}

__global__ void invden_k(float* den, int n) {
  int i = blockIdx.x * blockDim.x + threadIdx.x;
  if (i < n) den[i] = 1.f / fmaxf(den[i], 1e-9f);
}

__global__ void edge_agg_k(const int* esrc, const int* edst, const float* sbuf,
                           const float* invden, const float* zs, float* hout,
                           int E, int H, int out) {
  long long i = (long long)blockIdx.x * blockDim.x + threadIdx.x;
  if (i >= (long long)E * out) return;
  int e = (int)(i / out), d = (int)(i % out);
  int h = d / (out / H);
  float a = sbuf[(size_t)e * H + h] * invden[(size_t)edst[e] * H + h];
  atomicAdd(&hout[(size_t)edst[e] * out + d], a * zs[(size_t)esrc[e] * out + d]);
}

// ---------------- host-side param structs ----------------
struct LstmP { const float *b, *whh, *wih; };
struct GatP  { const float *ae, *al, *ar, *f1, *f2, *fb1, *fb2, *lb, *lg, *wd, *we, *ws; };
struct TfmP  { const float *f1, *f2, *fb1, *fb2, *lb1, *lb2, *lg1, *lg2, *wk, *wo, *wq, *wv; };

struct Bump {
  char* p; size_t off;
  float* f(size_t n)   { float* r = (float*)(p + off); off += ((n * 4) + 255) & ~(size_t)255; return r; }
  void*  raw(size_t b) { void*  r = (void*)(p + off);  off += (b + 255) & ~(size_t)255; return r; }
};

extern "C" void kernel_launch(void* const* d_in, const int* in_sizes, int n_in,
                              void* d_out, int out_size, void* d_ws, size_t ws_size,
                              hipStream_t stream) {
  (void)in_sizes; (void)n_in; (void)out_size; (void)ws_size;
  auto F = [&](int i) { return (const float*)d_in[i]; };
  auto I = [&](int i) { return (const int*)d_in[i]; };

  const int* word_id    = I(0);
  const int* sent_words = I(1);
  const int* sent_pos   = I(2);
  const int* ws_src = I(3); const int* ws_dst = I(4); const int* ws_tf = I(5);
  const int* ww_src = I(6); const int* ww_dst = I(7); const int* ww_tf = I(8);
  const int* ss_src = I(9); const int* ss_dst = I(10); const int* ss_tf = I(11);

  // params in jax pytree (sorted-key) order, indices 12..125
  int pi = 12;
  const float* cnn_proj_b = F(pi++);
  const float* cnn_proj_w = F(pi++);
  ConvArgs ca;
  for (int k = 0; k < 6; ++k) { ca.b[k] = F(pi++); ca.w[k] = F(pi++); }   // convs k2..k7 {b,w}
  const float* embed = F(pi++);
  auto rdL = [&]() { LstmP l; l.b = F(pi++); l.whh = F(pi++); l.wih = F(pi++); return l; };
  LstmP l0b = rdL(), l0f = rdL(), l1b = rdL(), l1f = rdL();               // lstm {l0b,l0f,l1b,l1f}
  const float* lstm_proj_b = F(pi++);
  const float* lstm_proj_w = F(pi++);
  LstmP lo0 = rdL(), lo1 = rdL();                                         // lstmout {l0,l1}
  const float* nfp_w = F(pi++);
  auto rdG = [&]() { GatP g; g.ae=F(pi++); g.al=F(pi++); g.ar=F(pi++); g.f1=F(pi++); g.f2=F(pi++);
                     g.fb1=F(pi++); g.fb2=F(pi++); g.lb=F(pi++); g.lg=F(pi++); g.wd=F(pi++);
                     g.we=F(pi++); g.ws=F(pi++); return g; };
  GatP gs2s = rdG(); GatP gs2w = rdG();
  const float* p_sim = F(pi++);
  const float* p_tf  = F(pi++);
  auto rdT = [&]() { TfmP t; t.f1=F(pi++); t.f2=F(pi++); t.fb1=F(pi++); t.fb2=F(pi++);
                     t.lb1=F(pi++); t.lb2=F(pi++); t.lg1=F(pi++); t.lg2=F(pi++);
                     t.wk=F(pi++); t.wo=F(pi++); t.wq=F(pi++); t.wv=F(pi++); return t; };
  TfmP tfm0 = rdT(), tfm1 = rdT();
  const float* tproj_b = F(pi++);
  const float* tproj_w = F(pi++);
  GatP gw2s = rdG(); GatP gw2w = rdG();
  const float* wh_b = F(pi++);
  const float* wh_w = F(pi++);

  // ---------- workspace ----------
  Bump ws{(char*)d_ws, 0};
  float* word_state = ws.f((size_t)cNW * 300);
  float* sent_state = ws.f((size_t)cNS * 64);
  float* sfeat      = ws.f((size_t)cNS * 64);
  size_t mark = ws.off;

  // ---------- phase 1: front end ----------
  float* pos51  = ws.f(51 * 300);
  float* pos128 = ws.f(50 * 128);
  float* ngram  = ws.f((size_t)cNS * 300);
  float* tmp300 = ws.f((size_t)cNS * 300);
  float* cat256 = ws.f((size_t)cNS * 256);
  float* xW     = ws.f((size_t)cNS * 512);
  float* h0     = ws.f((size_t)cNS * 256);
  float* h1     = ws.f((size_t)cNS * 256);
  float* tx     = ws.f((size_t)cNS * 128);
  float* qb     = ws.f((size_t)cNS * 128);
  float* kb     = ws.f((size_t)cNS * 128);
  float* vb     = ws.f((size_t)cNS * 128);
  float* ao     = ws.f((size_t)cNS * 128);
  float* ffn512 = ws.f((size_t)cNS * 512);
  _Float16* whh16 = (_Float16*)ws.raw(128 * 512 * 2);

  const size_t lstm_lds = 64 * 512 * 4 + 64 * 128 * 4 + 64 * 128 * 2;  // 176 KB dynamic LDS
  auto lstm_pass = [&](const float* X, int din, const LstmP& lp, float* hsout,
                       int ldh, int off, int rev) {
    f32_to_f16_k<<<(128 * 512 + 255) / 256, 256, 0, stream>>>(lp.whh, whh16, 128 * 512);
    gemm(stream, X, din, lp.wih, 512, lp.b, nullptr, xW, 512, cNS, 512, din, ACT_NONE);
    lstm_seq_k<<<1, 1024, lstm_lds, stream>>>(xW, whh16, hsout, ldh, off, rev);
  };

  sinusoid_k<<<(51 * 300 + 255) / 256, 256, 0, stream>>>(pos51, 51, 300, 1);
  sinusoid_k<<<(50 * 128 + 255) / 256, 256, 0, stream>>>(pos128, 50, 128, 0);
  gather_rows_k<<<((size_t)cNW * 300 + 255) / 256, 256, 0, stream>>>(embed, word_id, word_state, cNW, 300);
  conv_ngram_k<<<dim3(cNS, 6), 128, 0, stream>>>(sent_words, embed, ca, ngram);
  add_gather_pos_k<<<(cNS * 300 + 255) / 256, 256, 0, stream>>>(ngram, pos51, sent_pos, tmp300, cNS, 300);
  gemm(stream, tmp300, 300, cnn_proj_w, 128, cnn_proj_b, nullptr, cat256, 256, cNS, 128, 300, ACT_NONE);

  lstm_pass(ngram, 300, l0f, h0, 256, 0, 0);
  lstm_pass(ngram, 300, l0b, h0, 256, 128, 1);
  lstm_pass(h0, 256, l1f, h1, 256, 0, 0);
  lstm_pass(h0, 256, l1b, h1, 256, 128, 1);

  gemm(stream, h1, 256, lstm_proj_w, 128, lstm_proj_b, nullptr, tx, 128, cNS, 128, 256, ACT_NONE);
  add_pos_mod_k<<<(cNS * 128 + 255) / 256, 256, 0, stream>>>(tx, pos128, cNS, 128, 50);

  const TfmP* layers[2] = {&tfm0, &tfm1};
  for (int li = 0; li < 2; ++li) {
    const TfmP* L = layers[li];
    gemm(stream, tx, 128, L->wq, 128, nullptr, nullptr, qb, 128, cNS, 128, 128, ACT_NONE);
    gemm(stream, tx, 128, L->wk, 128, nullptr, nullptr, kb, 128, cNS, 128, 128, ACT_NONE);
    gemm(stream, tx, 128, L->wv, 128, nullptr, nullptr, vb, 128, cNS, 128, 128, ACT_NONE);
    attn_k<<<256, 128, 0, stream>>>(qb, kb, vb, ao);
    gemm(stream, ao, 128, L->wo, 128, nullptr, nullptr, qb, 128, cNS, 128, 128, ACT_NONE);
    ln_k<<<(cNS + 3) / 4, 128, 0, stream>>>(qb, tx, L->lg1, L->lb1, tx, cNS, 128);
    gemm(stream, tx, 128, L->f1, 512, L->fb1, nullptr, ffn512, 512, cNS, 512, 128, ACT_RELU);
    gemm(stream, ffn512, 512, L->f2, 128, L->fb2, tx, qb, 128, cNS, 128, 512, ACT_NONE);
    ln_k<<<(cNS + 3) / 4, 128, 0, stream>>>(qb, nullptr, L->lg2, L->lb2, tx, cNS, 128);
  }
  gemm(stream, tx, 128, tproj_w, 128, tproj_b, nullptr, cat256 + 128, 256, cNS, 128, 128, ACT_NONE);
  gemm(stream, cat256, 256, nfp_w, 64, nullptr, nullptr, sfeat, 64, cNS, 64, 256, ACT_NONE);

  // ---------- phase 2: GATs ----------
  auto do_gat = [&](const GatP& g, const float* hs, int dins, const float* hd, int dind,
                    const int* es, const int* ed, const int* et, const float* tftab,
                    int n_src, int n_dst, int out, int H, int E, float* nodeout) {
    ws.off = mark;
    float* zs   = ws.f((size_t)n_src * out);
    float* wsal = ws.f((size_t)dins * H);
    float* wdar = ws.f((size_t)dind * H);
    float* als  = ws.f((size_t)n_src * H);
    float* ard  = ws.f((size_t)n_dst * H);
    float* zea  = ws.f(10 * H);
    float* sbuf = ws.f((size_t)E * H);
    unsigned* mbuf = (unsigned*)ws.raw((size_t)n_dst * H * 4);
    float* den  = ws.f((size_t)n_dst * H);
    float* hout = ws.f((size_t)n_dst * out);
    float* ffn1 = ws.f((size_t)n_dst * 512);
    float* ybuf = ws.f((size_t)n_dst * out);

    fold_attw_k<<<(dins * H + 255) / 256, 256, 0, stream>>>(g.ws, g.al, wsal, dins, out, H);
    fold_attw_k<<<(dind * H + 255) / 256, 256, 0, stream>>>(g.wd, g.ar, wdar, dind, out, H);
    zea_k<<<1, 128, 0, stream>>>(tftab, g.we, g.ae, zea, out, H);
    gemm(stream, hs, dins, g.ws, out, nullptr, nullptr, zs, out, n_src, out, dins, ACT_NONE);
    gemm(stream, hs, dins, wsal, H, nullptr, nullptr, als, H, n_src, H, dins, ACT_NONE);
    gemm(stream, hd, dind, wdar, H, nullptr, nullptr, ard, H, n_dst, H, dind, ACT_NONE);
    fill_u32_k<<<(n_dst * H + 255) / 256, 256, 0, stream>>>(mbuf, 0x007FFFFFu, n_dst * H); // key(-inf)
    hipMemsetAsync(den, 0, (size_t)n_dst * H * 4, stream);
    hipMemsetAsync(hout, 0, (size_t)n_dst * out * 4, stream);
    edge_score_k<<<(E * H + 255) / 256, 256, 0, stream>>>(es, ed, et, als, ard, zea, sbuf, mbuf, E, H);
    edge_exp_k<<<(E * H + 255) / 256, 256, 0, stream>>>(ed, mbuf, sbuf, den, E, H);
    invden_k<<<(n_dst * H + 255) / 256, 256, 0, stream>>>(den, n_dst * H);
    long long tot = (long long)E * out;
    edge_agg_k<<<(unsigned)((tot + 255) / 256), 256, 0, stream>>>(es, ed, sbuf, den, zs, hout, E, H, out);
    elu_k<<<(n_dst * out + 255) / 256, 256, 0, stream>>>(hout, n_dst * out);
    gemm(stream, hout, out, g.f1, 512, g.fb1, nullptr, ffn1, 512, n_dst, 512, out, ACT_RELU);
    gemm(stream, ffn1, 512, g.f2, out, g.fb2, hout, ybuf, out, n_dst, out, 512, ACT_NONE);
    ln_k<<<(n_dst + 3) / 4, 128, 0, stream>>>(ybuf, nullptr, g.lg, g.lb, nodeout, n_dst, out);
  };

  do_gat(gw2s, word_state, 300, sfeat, 64, ws_src, ws_dst, ws_tf, p_tf,
         cNW, cNS, 64, 8, cEWS, sent_state);
  for (int it = 0; it < 2; ++it) {
    do_gat(gw2w, word_state, 300, word_state, 300, ww_src, ww_dst, ww_tf, p_tf,
           cNW, cNW, 300, 6, cEWW, word_state);
    do_gat(gs2w, sent_state, 64, word_state, 300, ws_dst, ws_src, ws_tf, p_tf,
           cNS, cNW, 300, 6, cEWS, word_state);
    do_gat(gw2s, word_state, 300, sent_state, 64, ws_src, ws_dst, ws_tf, p_tf,
           cNW, cNS, 64, 8, cEWS, sent_state);
    do_gat(gs2s, sent_state, 64, sent_state, 64, ss_src, ss_dst, ss_tf, p_sim,
           cNS, cNS, 64, 8, cESS, sent_state);
  }

  // ---------- phase 3: output LSTMs + head ----------
  ws.off = mark;
  xW = ws.f((size_t)cNS * 512);
  whh16 = (_Float16*)ws.raw(128 * 512 * 2);
  float* y0 = ws.f((size_t)cNS * 128);
  float* y1 = ws.f((size_t)cNS * 128);

  f32_to_f16_k<<<(128 * 512 + 255) / 256, 256, 0, stream>>>(lo0.whh, whh16, 128 * 512);
  gemm(stream, sent_state, 64, lo0.wih, 512, lo0.b, nullptr, xW, 512, cNS, 512, 64, ACT_NONE);
  lstm_seq_k<<<1, 1024, lstm_lds, stream>>>(xW, whh16, y0, 128, 0, 0);
  f32_to_f16_k<<<(128 * 512 + 255) / 256, 256, 0, stream>>>(lo1.whh, whh16, 128 * 512);
  gemm(stream, y0, 128, lo1.wih, 512, lo1.b, nullptr, xW, 512, cNS, 512, 128, ACT_NONE);
  lstm_seq_k<<<1, 1024, lstm_lds, stream>>>(xW, whh16, y1, 128, 0, 0);

  gemm(stream, y1, 128, wh_w, 2, wh_b, nullptr, (float*)d_out, 2, cNS, 2, 128, ACT_SIG);
}
